// SegmentCompressor_8272107012360
// MI455X (gfx1250) — matrix operand order repaired and checked by reading, not compiled, not run on hardware
//
#include <hip/hip_runtime.h>
#include <math.h>

#define HIDDEN 1024
#define SLOTS  16
#define BSZ    512
#define SEG    256
#define SCALE  0.03125f  /* 1/sqrt(1024) */

typedef unsigned short ushort_t;
typedef __attribute__((ext_vector_type(16))) __bf16 v16bf;
typedef __attribute__((ext_vector_type(8)))  __bf16 v8bf;
typedef __attribute__((ext_vector_type(8)))  float  v8f;

union AU { v16bf v; v8bf h[2]; __bf16 e[16]; };
union CU { v8f v; float f[8]; };

__device__ __forceinline__ v8f wmma_bf16(v16bf a, v16bf b, v8f c) {
  // D = A(16x32 bf16) * B(32x16 bf16) + C(16x16 f32)
  return __builtin_amdgcn_wmma_f32_16x16x32_bf16(false, a, false, b,
                                                 (short)0, c, false, false);
}

// ---------------------------------------------------------------------------
// K0: query[s][e] = latent[s]·Wq[e] + bq[e]          (16 x 1024, fp32)
// ---------------------------------------------------------------------------
__global__ __launch_bounds__(256) void qproj_kernel(
    const float* __restrict__ latent, const float* __restrict__ Wq,
    const float* __restrict__ bq, float* __restrict__ query) {
  int idx = blockIdx.x * 256 + threadIdx.x;     // 16384 threads
  int s = idx >> 10, e = idx & 1023;
  const float* lr = latent + (size_t)s * HIDDEN;
  const float* wr = Wq + (size_t)e * HIDDEN;
  float acc = 0.f;
  for (int d = 0; d < HIDDEN; ++d) acc += lr[d] * wr[d];
  query[idx] = acc + bq[e];
}

// ---------------------------------------------------------------------------
// K1: qk[s][d] = scale * (query[s]·Wk[:,d])   (bf16, pre-scaled)
//     qkb[s]   = scale * (query[s]·bk)
// ---------------------------------------------------------------------------
__global__ __launch_bounds__(256) void qkproj_kernel(
    const float* __restrict__ query, const float* __restrict__ Wk,
    const float* __restrict__ bk, ushort_t* __restrict__ qk_u,
    float* __restrict__ qkb) {
  int idx = blockIdx.x * 256 + threadIdx.x;     // 16384 threads
  int s = idx >> 10, d = idx & 1023;
  const float* qr = query + (size_t)s * HIDDEN;
  float acc = 0.f;
  for (int e = 0; e < HIDDEN; ++e) acc += qr[e] * Wk[(size_t)e * HIDDEN + d];
  ((__bf16*)qk_u)[idx] = (__bf16)(acc * SCALE);
  if (d == 0) {
    float b = 0.f;
    for (int e = 0; e < HIDDEN; ++e) b += qr[e] * bk[e];
    qkb[s] = b * SCALE;
  }
}

// ---------------------------------------------------------------------------
// K3: bvo[e] = Wo[e]·bv + bo[e]
// ---------------------------------------------------------------------------
__global__ __launch_bounds__(256) void bvo_kernel(
    const float* __restrict__ Wo, const float* __restrict__ bv,
    const float* __restrict__ bo, float* __restrict__ bvo) {
  int e = blockIdx.x * 256 + threadIdx.x;       // 1024 threads
  const float* wr = Wo + (size_t)e * HIDDEN;
  float acc = 0.f;
  for (int f = 0; f < HIDDEN; ++f) acc += wr[f] * bv[f];
  bvo[e] = acc + bo[e];
}

// ---------------------------------------------------------------------------
// K2: Wov[e][d] = sum_f Wo[e][f] * Wv[f][d], stored pre-swizzled as the
// WMMA B-operand layout Wb[kc][nt][lane][16]:
//   element (d,e):  kc=d>>5, half=(d>>4)&1, lane=16*half+(e&15), slot=d&15
// WMMA tiling: A = Wo rows (M=e), B = Wv (K=f, N=d). One 16x16 tile / wave.
// ---------------------------------------------------------------------------
__global__ __launch_bounds__(256) void wov_kernel(
    const float* __restrict__ Wo, const float* __restrict__ Wv,
    ushort_t* __restrict__ Wb_u) {
  __bf16* Wb = (__bf16*)Wb_u;
  int w = threadIdx.x >> 5, l = threadIdx.x & 31;
  int n = l & 15, half = l >> 4;
  int tile = blockIdx.x * 8 + w;                // 4096 tiles total
  int e0 = (tile & 63) << 4;                    // M tile (e)
  int d0 = (tile >> 6) << 4;                    // N tile (d)

  v8f c = {};
  for (int k0 = 0; k0 < HIDDEN; k0 += 32) {
    AU a, b;
    const float* ar = Wo + (size_t)(e0 + n) * HIDDEN + k0;
#pragma unroll
    for (int j = 0; j < 8; ++j) {
      a.e[j]     = (__bf16)ar[16 * half + j];
      a.e[8 + j] = (__bf16)ar[16 * half + 8 + j];
    }
    const float* br = Wv + (size_t)(k0 + 16 * half) * HIDDEN + d0 + n;
#pragma unroll
    for (int j = 0; j < 16; ++j) b.e[j] = (__bf16)br[(size_t)j * HIDDEN];
    c = wmma_bf16(a.v, b.v, c);
  }

  // Pre-swizzled store: lane has fixed d = d0+n, e = e0 + r + 8*half
  int d = d0 + n;
  int kc = d >> 5, hw = (d >> 4) & 1, slot = d & 15;
  __bf16* base = Wb + ((size_t)(kc * 64 + (e0 >> 4)) * 32) * 16;
  CU cu; cu.v = c;
#pragma unroll
  for (int r = 0; r < 8; ++r) {
    int lanep = 16 * hw + r + 8 * half;
    base[(size_t)lanep * 16 + slot] = (__bf16)cu.f[r];
  }
}

// ---------------------------------------------------------------------------
// Main fused kernel: one workgroup (8 wave32) per batch element.
//   Pass 1: scores[16][256] = qk(bf16) @ X[b]^T        (WMMA)
//   softmax over t (+qkb, +logits)  -> attn bf16 in LDS
//   Pass 2: Y[16][1024] = attn @ X[b]                  (WMMA) -> LDS bf16
//   Pass 3: out[b] = Y @ Wb + bvo                      (WMMA)
// ---------------------------------------------------------------------------
__global__ __launch_bounds__(256) void seg_attn_kernel(
    const float* __restrict__ X, const float* __restrict__ logits,
    const ushort_t* __restrict__ qk_u, const float* __restrict__ qkb,
    const ushort_t* __restrict__ Wb_u, const float* __restrict__ bvo,
    float* __restrict__ out) {
  __shared__ float  sc[SLOTS][SEG];      // 16 KB
  __shared__ __bf16 attn[SLOTS][SEG];    //  8 KB
  __shared__ __bf16 Ybf[SLOTS][HIDDEN];  // 32 KB

  const __bf16* qk = (const __bf16*)qk_u;
  const __bf16* Wb = (const __bf16*)Wb_u;

  int b = blockIdx.x;
  const float* Xb = X + (size_t)b * SEG * HIDDEN;
  int tid = threadIdx.x, w = tid >> 5, l = tid & 31;
  int n = l & 15, half = l >> 4;

  // ---------------- Pass 1: scores = qk @ X^T --------------------------------
  for (int tt = w; tt < 16; tt += 8) {   // each wave owns 2 t-tiles
    int t0 = tt * 16;
    v8f c = {};
    for (int k0 = 0; k0 < HIDDEN; k0 += 32) {
      AU a, bm;
      const __bf16* ap = qk + (size_t)n * HIDDEN + k0;
      a.h[0] = *(const v8bf*)(ap + 16 * half);
      a.h[1] = *(const v8bf*)(ap + 16 * half + 8);
      const float* bp = Xb + (size_t)(t0 + n) * HIDDEN + k0 + 16 * half;
      __builtin_prefetch(bp + 256, 0, 1);
#pragma unroll
      for (int j = 0; j < 16; ++j) bm.e[j] = (__bf16)bp[j];
      c = wmma_bf16(a.v, bm.v, c);
    }
    CU cu; cu.v = c;
#pragma unroll
    for (int r = 0; r < 8; ++r) sc[r + 8 * half][t0 + n] = cu.f[r];
  }
  __syncthreads();

  // ---------------- Softmax over t ------------------------------------------
  for (int s = w; s < SLOTS; s += 8) {   // 2 rows per wave
    float qb = qkb[s];
    float vals[8];
    float mx = -1e30f;
#pragma unroll
    for (int i = 0; i < 8; ++i) {
      int t = l + 32 * i;
      float v = sc[s][t] + qb + logits[(size_t)b * SEG + t];
      vals[i] = v;
      mx = fmaxf(mx, v);
    }
#pragma unroll
    for (int o = 16; o > 0; o >>= 1) mx = fmaxf(mx, __shfl_xor(mx, o, 32));
    float sum = 0.f;
#pragma unroll
    for (int i = 0; i < 8; ++i) { vals[i] = __expf(vals[i] - mx); sum += vals[i]; }
#pragma unroll
    for (int o = 16; o > 0; o >>= 1) sum += __shfl_xor(sum, o, 32);
    float inv = 1.0f / sum;
#pragma unroll
    for (int i = 0; i < 8; ++i) attn[s][l + 32 * i] = (__bf16)(vals[i] * inv);
  }
  __syncthreads();

  // ---------------- Pass 2: Y = attn @ X ------------------------------------
  for (int i = 0; i < 8; ++i) {          // 8 d-tiles per wave
    int d0 = (w + 8 * i) * 16;
    v8f c = {};
    for (int k0 = 0; k0 < SEG; k0 += 32) {
      AU a, bm;
      const __bf16* ap = &attn[n][k0];
      a.h[0] = *(const v8bf*)(ap + 16 * half);
      a.h[1] = *(const v8bf*)(ap + 16 * half + 8);
      const float* bp = Xb + (size_t)(k0 + 16 * half) * HIDDEN + d0 + n;
#pragma unroll
      for (int j = 0; j < 16; ++j) bm.e[j] = (__bf16)bp[(size_t)j * HIDDEN];
      c = wmma_bf16(a.v, bm.v, c);
    }
    CU cu; cu.v = c;
#pragma unroll
    for (int r = 0; r < 8; ++r) Ybf[r + 8 * half][d0 + n] = (__bf16)cu.f[r];
  }
  __syncthreads();

  // ---------------- Pass 3: out = Y @ Wb + bvo ------------------------------
  float* outb = out + (size_t)b * SLOTS * HIDDEN;
  for (int i = 0; i < 8; ++i) {          // 8 e-tiles per wave
    int nt = w + 8 * i;
    int e0 = nt * 16;
    v8f c = {};
    for (int kc = 0; kc < 32; ++kc) {
      AU a;
      const __bf16* ap = &Ybf[n][kc * 32];
      a.h[0] = *(const v8bf*)(ap + 16 * half);
      a.h[1] = *(const v8bf*)(ap + 16 * half + 8);
      // Pre-swizzled B: 32 contiguous bytes per lane.
      v16bf bv16 = *(const v16bf*)(Wb + ((size_t)(kc * 64 + nt) * 32 + l) * 16);
      c = wmma_bf16(a.v, bv16, c);
    }
    float bb = bvo[e0 + n];
    CU cu; cu.v = c;
#pragma unroll
    for (int r = 0; r < 8; ++r)
      outb[(size_t)(r + 8 * half) * HIDDEN + e0 + n] = cu.f[r] + bb;
  }
}

// ---------------------------------------------------------------------------
// Workspace layout (bytes):
//   [0)        query   fp32 16*1024        = 65536
//   [65536)    qk      bf16 16*1024        = 32768
//   [98304)    qkb     fp32 16             (pad)
//   [98816)    bvo     fp32 1024           = 4096
//   [103424)   Wb      bf16 32*64*32*16    = 2 MiB   (32B aligned)
// ---------------------------------------------------------------------------
extern "C" void kernel_launch(void* const* d_in, const int* in_sizes, int n_in,
                              void* d_out, int out_size, void* d_ws, size_t ws_size,
                              hipStream_t stream) {
  const float* X      = (const float*)d_in[0];   // segment_states [512,256,1024]
  const float* logits = (const float*)d_in[1];   // importance_logits [512,256]
  const float* latent = (const float*)d_in[2];   // latent_queries [16,1024]
  const float* Wq     = (const float*)d_in[3];
  const float* bq     = (const float*)d_in[4];
  const float* Wk     = (const float*)d_in[5];
  const float* bk     = (const float*)d_in[6];
  const float* Wv     = (const float*)d_in[7];
  const float* bv     = (const float*)d_in[8];
  const float* Wo     = (const float*)d_in[9];
  const float* bo     = (const float*)d_in[10];
  float* out = (float*)d_out;

  char* ws = (char*)d_ws;
  float*    query = (float*)(ws + 0);
  ushort_t* qk    = (ushort_t*)(ws + 65536);
  float*    qkb   = (float*)(ws + 98304);
  float*    bvo   = (float*)(ws + 98816);
  ushort_t* Wb    = (ushort_t*)(ws + 103424);

  qproj_kernel<<<64, 256, 0, stream>>>(latent, Wq, bq, query);
  qkproj_kernel<<<64, 256, 0, stream>>>(query, Wk, bk, qk, qkb);
  bvo_kernel<<<4, 256, 0, stream>>>(Wo, bv, bo, bvo);
  wov_kernel<<<512, 256, 0, stream>>>(Wo, Wv, Wb);
  seg_attn_kernel<<<BSZ, 256, 0, stream>>>(X, logits, qk, qkb, Wb, bvo, out);
}
